// DHSLayer_90933047591134
// MI455X (gfx1250) — matrix operand order repaired and checked by reading, not compiled, not run on hardware
//
#include <hip/hip_runtime.h>
#include <cstdint>

typedef __attribute__((ext_vector_type(16))) _Float16 v16h;
typedef __attribute__((ext_vector_type(8)))  _Float16 v8h;
typedef __attribute__((ext_vector_type(8)))  float    v8f;

#define S_LEN 197
#define SP    224      // S padded to 14*16
#define DMODEL 768
#define NHEAD 12
#define DHEAD 64
#define BATCH 128

__device__ __forceinline__ v16h make_a(const _Float16* lo, const _Float16* hi) {
  v8h l = *(const v8h*)lo;
  v8h h = *(const v8h*)hi;
  v16h r;
#pragma unroll
  for (int i = 0; i < 8; ++i) { r[i] = l[i]; r[i + 8] = h[i]; }
  return r;
}

__device__ __forceinline__ v8f wmma_f16(v16h a, v16h b, v8f c) {
  return __builtin_amdgcn_wmma_f32_16x16x32_f16(false, a, false, b, (short)0, c,
                                                false, false);
}

// CDNA5 async copy global->LDS (ASYNCcnt-tracked), GV addressing mode.
__device__ __forceinline__ void async_load_b128(unsigned lds_off,
                                                unsigned long long gaddr) {
  asm volatile("global_load_async_to_lds_b128 %0, %1, off"
               :
               : "v"(lds_off), "v"(gaddr)
               : "memory");
}

// Leading barrier: my async parts for the current buffer have landed (wait
// asynccnt<=1 keeps the next prefetch in flight), then rendezvous.
#define BAR_ASYNC_1() \
  asm volatile("s_wait_asynccnt 0x1\ns_barrier_signal -1\ns_barrier_wait -1" ::: "memory")
#define BAR_ASYNC_0() \
  asm volatile("s_wait_asynccnt 0x0\ns_barrier_signal -1\ns_barrier_wait -1" ::: "memory")
// Trailing barrier: my LDS reads retired (dscnt==0) before anyone overwrites.
#define BAR_DS() \
  asm volatile("s_wait_dscnt 0x0\ns_barrier_signal -1\ns_barrier_wait -1" ::: "memory")

// 4 WMMAs of one 16x64 tile against a staged 64n x 32k LDS B-tile.
__device__ __forceinline__ void gemm_step(v16h a, const _Float16* bbuf, int nl,
                                          int gg, v8f acc[4]) {
#pragma unroll
  for (int t2 = 0; t2 < 4; ++t2) {
    v16h bm = make_a(bbuf + (t2 * 16 + nl) * 32 + 8 * gg, bbuf + (t2 * 16 + nl) * 32 + 16 + 8 * gg);
    acc[t2] = wmma_f16(a, bm, acc[t2]);
  }
}

// ---------------- weight transpose + f16 convert: Wt[n*768+k] = W[k*768+n] ---
__global__ void convert_w(const float* Wq, const float* Wk, const float* Wv,
                          const float* Wd, _Float16* out) {
  int z = blockIdx.y;
  const float* src = (z == 0) ? Wq : (z == 1) ? Wk : (z == 2) ? Wv : Wd;
  _Float16* dst = out + (size_t)z * DMODEL * DMODEL;
  int idx = blockIdx.x * 256 + threadIdx.x;       // n*768 + k
  int n = idx / DMODEL, k = idx % DMODEL;
  dst[idx] = (_Float16)src[k * DMODEL + n];
}

// ---------------- hidden fp32 -> f16, zero-padded rows to SP -----------------
__global__ void convert_x(const float* hid, _Float16* xb) {
  size_t idx = (size_t)blockIdx.x * 256 + threadIdx.x;  // (b*SP+s)*768+k
  int kcol = (int)(idx % DMODEL);
  size_t row = idx / DMODEL;
  int s = (int)(row % SP);
  int b = (int)(row / SP);
  _Float16 v = (_Float16)0.0f;
  if (s < S_LEN) v = (_Float16)hid[((size_t)b * S_LEN + s) * DMODEL + kcol];
  xb[idx] = v;
}

// ---------------- DHS scoring MLP -> mask bytes ------------------------------
__global__ void score_mask(const float* hid, const float* W1, const float* b1,
                           const float* W2, const float* b2,
                           unsigned char* mask) {
  int b = blockIdx.x / (S_LEN - 1);
  int s = 1 + blockIdx.x % (S_LEN - 1);
  int j = threadIdx.x;  // 0..63
  const float* row = hid + ((size_t)b * S_LEN + s) * DMODEL;
  float acc = 0.f;
  for (int k = 0; k < DMODEL; ++k) acc = fmaf(row[k], W1[k * 64 + j], acc);
  float hr = fmaxf(acc + b1[j], 0.f);
  __shared__ float sm[64];
  sm[j] = hr * W2[j];
  __syncthreads();
  for (int off = 32; off > 0; off >>= 1) {
    if (j < off) sm[j] += sm[j + off];
    __syncthreads();
  }
  if (j == 0) {
    float sig = 1.f / (1.f + __expf(-(sm[0] + b2[0])));
    mask[b * S_LEN + s] = (sig >= 0.05f) ? 1 : 0;
  }
  if (j == 0 && blockIdx.x < BATCH) mask[blockIdx.x * S_LEN] = 1;  // CLS kept
}

// ---------------- fused QKV projection GEMM ---------------------------------
// Block = 8 waves sharing one (g, n64): B tile (64n x 32k) double-buffered in
// LDS via global_load_async_to_lds_b128; each wave owns a 16x64 output tile.
__global__ void qkv_gemm(const _Float16* xb, const _Float16* wt,
                         const float* bq, const float* bk, const float* bv,
                         _Float16* qb, _Float16* kb, _Float16* vtb) {
  __shared__ _Float16 bstage[2][64][32];  // 2 x 4KB
  int tid = threadIdx.x;
  int wid = tid >> 5, lane = tid & 31;
  int bx = blockIdx.x;
  int mblk = bx % 224;
  int n64 = (bx / 224) % 12;              // == head index
  int g = bx / (224 * 12);                // 0=q 1=k 2=v
  int mTile = mblk * 8 + wid;
  int b = mTile / (SP / 16), t = mTile % (SP / 16);
  int nl = lane & 15, gg = lane >> 4;

  const _Float16* w = wt + (size_t)g * DMODEL * DMODEL;
  // per-thread async stage: 256 threads x 16B = 4KB = 64 rows x 64B
  int rr = tid >> 2, cc = tid & 3;
  unsigned long long gsrc =
      (unsigned long long)(uintptr_t)(w + (size_t)(n64 * 64 + rr) * DMODEL +
                                      cc * 8);
  unsigned ldst = (unsigned)(uintptr_t)&bstage[0][rr][cc * 8];
  const _Float16* bb0 = &bstage[0][0][0];

  const _Float16* arow = xb + ((size_t)b * SP + t * 16 + nl) * DMODEL;
  v8f acc[4] = {};

  async_load_b128(ldst, gsrc);  // kk=0 -> buf0
#pragma unroll 1
  for (int kki = 0; kki < 23; ++kki) {
    async_load_b128(ldst + ((unsigned)((kki + 1) & 1) << 12),
                    gsrc + (unsigned long long)(kki + 1) * 64);
    v16h a = make_a(arow + kki * 32 + 8 * gg, arow + kki * 32 + 16 + 8 * gg);
    BAR_ASYNC_1();
    gemm_step(a, bb0 + ((kki & 1) << 11), nl, gg, acc);
    BAR_DS();
  }
  {  // peeled final k-step (buffer 1)
    v16h a = make_a(arow + 23 * 32 + 8 * gg, arow + 23 * 32 + 16 + 8 * gg);
    BAR_ASYNC_0();
    gemm_step(a, bb0 + (1 << 11), nl, gg, acc);
  }

  const float* bias = (g == 0) ? bq : (g == 1) ? bk : bv;
  size_t bh = (size_t)b * NHEAD + n64;
#pragma unroll
  for (int t2 = 0; t2 < 4; ++t2) {
    int n = n64 * 64 + t2 * 16 + nl;
    int dh = t2 * 16 + nl;
    float bi = bias[n];
#pragma unroll
    for (int r = 0; r < 8; ++r) {
      int s = t * 16 + r + 8 * gg;
      _Float16 v = (_Float16)(acc[t2][r] + bi);
      if (g == 0)      qb[(bh * SP + s) * DHEAD + dh] = v;
      else if (g == 1) kb[(bh * SP + s) * DHEAD + dh] = v;
      else             vtb[(bh * DHEAD + dh) * SP + s] = v;  // V transposed
    }
  }
}

// ---------------- flash attention: one wave = (b,h,16-query tile) ------------
__global__ void attention(const _Float16* qb, const _Float16* kb,
                          const _Float16* vtb, _Float16* ctx) {
  int bh = blockIdx.x;
  int b = bh / NHEAD, h = bh % NHEAD;
  int wid = threadIdx.x / 32;
  int lane = threadIdx.x % 32;
  int qTile = blockIdx.y * 4 + wid;
  if (qTile >= 13) return;  // ceil(197/16); waves independent, no block sync
  int nl = lane & 15, gg = lane >> 4;
  __shared__ _Float16 pst[4][16][40];  // 80B row stride -> 16B-aligned reloads

  const _Float16* qrow = qb + ((size_t)bh * SP + qTile * 16 + nl) * DHEAD;
  v16h a0 = make_a(qrow + 8 * gg, qrow + 16 + 8 * gg);
  v16h a1 = make_a(qrow + 32 + 8 * gg, qrow + 48 + 8 * gg);
  v8f acc[4] = {};
  float rmax[8], rsum[8];
#pragma unroll
  for (int r = 0; r < 8; ++r) { rmax[r] = -3.0e38f; rsum[r] = 0.f; }

  const _Float16* kbase = kb + (size_t)bh * SP * DHEAD;
  const _Float16* vbase0 = vtb + (size_t)bh * DHEAD * SP;

  for (int kc = 0; kc < SP / 32; ++kc) {
    int key0 = kc * 32 + nl;
    int key1 = key0 + 16;
    v16h b00 = make_a(kbase + (size_t)key0 * DHEAD + 0 + 8 * gg, kbase + (size_t)key0 * DHEAD + 16 + 8 * gg);
    v16h b01 = make_a(kbase + (size_t)key0 * DHEAD + 32 + 8 * gg, kbase + (size_t)key0 * DHEAD + 48 + 8 * gg);
    v16h b10 = make_a(kbase + (size_t)key1 * DHEAD + 0 + 8 * gg, kbase + (size_t)key1 * DHEAD + 16 + 8 * gg);
    v16h b11 = make_a(kbase + (size_t)key1 * DHEAD + 32 + 8 * gg, kbase + (size_t)key1 * DHEAD + 48 + 8 * gg);
    v8f s0 = {};
    s0 = wmma_f16(a0, b00, s0);
    s0 = wmma_f16(a1, b01, s0);
    v8f s1 = {};
    s1 = wmma_f16(a0, b10, s1);
    s1 = wmma_f16(a1, b11, s1);
    bool ok0 = key0 < S_LEN, ok1 = key1 < S_LEN;
#pragma unroll
    for (int r = 0; r < 8; ++r) {
      float x0 = ok0 ? s0[r] * 0.125f : -1.0e30f;  // 1/sqrt(64)
      float x1 = ok1 ? s1[r] * 0.125f : -1.0e30f;
      float mx = fmaxf(x0, x1);
      for (int d = 1; d < 16; d <<= 1) mx = fmaxf(mx, __shfl_xor(mx, d, 32));
      float nm = fmaxf(rmax[r], mx);
      float sc = __expf(rmax[r] - nm);
      rmax[r] = nm;
      float p0 = __expf(x0 - nm);
      float p1 = __expf(x1 - nm);
      float ps = p0 + p1;
      for (int d = 1; d < 16; d <<= 1) ps += __shfl_xor(ps, d, 32);
      rsum[r] = rsum[r] * sc + ps;
#pragma unroll
      for (int t2 = 0; t2 < 4; ++t2) acc[t2][r] *= sc;
      int mrow = r + 8 * gg;
      pst[wid][mrow][nl] = (_Float16)p0;
      pst[wid][mrow][16 + nl] = (_Float16)p1;
    }
    asm volatile("s_wait_dscnt 0" ::: "memory");  // LDS RAW (cross-lane stage)
    const _Float16* prow = &pst[wid][nl][0];
    v16h ap = make_a(prow + 8 * gg, prow + 16 + 8 * gg);
    const _Float16* vb = vbase0 + kc * 32 + 8 * gg;
#pragma unroll
    for (int t2 = 0; t2 < 4; ++t2) {
      int dh = t2 * 16 + nl;
      v16h bv = make_a(vb + (size_t)dh * SP, vb + (size_t)dh * SP + 16);
      acc[t2] = wmma_f16(ap, bv, acc[t2]);
    }
  }
#pragma unroll
  for (int t2 = 0; t2 < 4; ++t2) {
#pragma unroll
    for (int r = 0; r < 8; ++r) {
      int srow = qTile * 16 + r + 8 * gg;
      float inv = 1.f / rsum[r];
      ctx[((size_t)b * SP + srow) * DMODEL + h * DHEAD + t2 * 16 + nl] =
          (_Float16)(acc[t2][r] * inv);
    }
  }
}

// ---------------- dense + bias + residual + masked scatter -------------------
// Same async-LDS double-buffered B staging as qkv_gemm.
__global__ void dense_gemm(const _Float16* ctx, const _Float16* wdt,
                           const float* bd, const unsigned char* mask,
                           const float* hid, float* out) {
  __shared__ _Float16 bstage[2][64][32];
  int tid = threadIdx.x;
  int wid = tid >> 5, lane = tid & 31;
  int bx = blockIdx.x;
  int mblk = bx % 208;
  int n64 = bx / 208;  // 0..11
  int mTile = mblk * 8 + wid;
  int b = mTile / 13, t = mTile % 13;
  int nl = lane & 15, gg = lane >> 4;

  int rr = tid >> 2, cc = tid & 3;
  unsigned long long gsrc =
      (unsigned long long)(uintptr_t)(wdt + (size_t)(n64 * 64 + rr) * DMODEL +
                                      cc * 8);
  unsigned ldst = (unsigned)(uintptr_t)&bstage[0][rr][cc * 8];
  const _Float16* bb0 = &bstage[0][0][0];

  const _Float16* arow = ctx + ((size_t)b * SP + t * 16 + nl) * DMODEL;
  v8f acc[4] = {};

  async_load_b128(ldst, gsrc);
#pragma unroll 1
  for (int kki = 0; kki < 23; ++kki) {
    async_load_b128(ldst + ((unsigned)((kki + 1) & 1) << 12),
                    gsrc + (unsigned long long)(kki + 1) * 64);
    v16h a = make_a(arow + kki * 32 + 8 * gg, arow + kki * 32 + 16 + 8 * gg);
    BAR_ASYNC_1();
    gemm_step(a, bb0 + ((kki & 1) << 11), nl, gg, acc);
    BAR_DS();
  }
  {
    v16h a = make_a(arow + 23 * 32 + 8 * gg, arow + 23 * 32 + 16 + 8 * gg);
    BAR_ASYNC_0();
    gemm_step(a, bb0 + (1 << 11), nl, gg, acc);
  }

#pragma unroll
  for (int t2 = 0; t2 < 4; ++t2) {
    int n = n64 * 64 + t2 * 16 + nl;
    float bi = bd[n];
#pragma unroll
    for (int r = 0; r < 8; ++r) {
      int s = t * 16 + r + 8 * gg;
      if (s < S_LEN) {
        size_t gi = ((size_t)b * S_LEN + s) * DMODEL + n;
        float res = (float)ctx[((size_t)b * SP + s) * DMODEL + n];
        float val = acc[t2][r] + bi + res;
        out[gi] = mask[b * S_LEN + s] ? val : hid[gi];
      }
    }
  }
}

extern "C" void kernel_launch(void* const* d_in, const int* in_sizes, int n_in,
                              void* d_out, int out_size, void* d_ws,
                              size_t ws_size, hipStream_t stream) {
  const float* hid = (const float*)d_in[0];
  const float* Wq = (const float*)d_in[1];
  const float* bq = (const float*)d_in[2];
  const float* Wk = (const float*)d_in[3];
  const float* bk = (const float*)d_in[4];
  const float* Wv = (const float*)d_in[5];
  const float* bv = (const float*)d_in[6];
  const float* Wd = (const float*)d_in[7];
  const float* bd = (const float*)d_in[8];
  const float* W1 = (const float*)d_in[9];
  const float* b1 = (const float*)d_in[10];
  const float* W2 = (const float*)d_in[11];
  const float* b2 = (const float*)d_in[12];
  float* out = (float*)d_out;

  char* ws = (char*)d_ws;
  size_t off = 0;
  _Float16* wt = (_Float16*)(ws);                    // 4 transposed weights
  off += (size_t)4 * DMODEL * DMODEL * 2;
  size_t xbsz = (size_t)BATCH * SP * DMODEL * 2;     // 44 MB each
  _Float16* xb = (_Float16*)(ws + off); off += xbsz;
  _Float16* qb = (_Float16*)(ws + off); off += xbsz;
  _Float16* kb = (_Float16*)(ws + off); off += xbsz;
  _Float16* vtb = (_Float16*)(ws + off); off += xbsz;
  unsigned char* mask = (unsigned char*)(ws + off);
  _Float16* ctx = xb;  // alias: xb dead after qkv_gemm (stream-ordered)

  convert_w<<<dim3(DMODEL * DMODEL / 256, 4), 256, 0, stream>>>(Wq, Wk, Wv, Wd,
                                                                wt);
  convert_x<<<dim3((unsigned)(xbsz / 2 / 256)), 256, 0, stream>>>(hid, xb);
  score_mask<<<dim3(BATCH * (S_LEN - 1)), 64, 0, stream>>>(hid, W1, b1, W2, b2,
                                                           mask);
  qkv_gemm<<<dim3(3 * 12 * 224), 256, 0, stream>>>(xb, wt, bq, bk, bv, qb, kb,
                                                   vtb);
  attention<<<dim3(BATCH * NHEAD, 4), 128, 0, stream>>>(qb, kb, vtb, ctx);
  dense_gemm<<<dim3(12 * 208), 256, 0, stream>>>(
      ctx, wt + (size_t)3 * DMODEL * DMODEL, bd, mask, hid, out);
}
